// BidirectionalAttn_8211977470059
// MI455X (gfx1250) — matrix operand order repaired and checked by reading, not compiled, not run on hardware
//
#include <hip/hip_runtime.h>
#include <hip/hip_bf16.h>

// ---------------------------------------------------------------------------
// BidirectionalAttn for MI455X (gfx1250, wave32, WMMA + TDM + async-LDS).
// bf16 WMMA (f32 accumulate) everywhere; flash-style fused dual-softmax
// attention so the 512MB logits tensor is never materialized.
// GEMM tiles staged by the Tensor Data Mover (double buffered); attention
// K-tiles staged with global_load_async_to_lds_b128.
// ---------------------------------------------------------------------------

typedef __bf16 bf16;
typedef __attribute__((ext_vector_type(16))) __bf16 v16bf;
typedef __attribute__((ext_vector_type(8)))  __bf16 v8bf;
typedef __attribute__((ext_vector_type(8)))  float  v8f;
typedef __attribute__((ext_vector_type(4)))  unsigned int u32x4;
typedef __attribute__((ext_vector_type(8)))  int i32x8;
typedef __attribute__((ext_vector_type(4)))  int i32x4;

union Frag16 { v16bf v; v8bf h[2]; };

__device__ __forceinline__ v8f zero8() {
  v8f v;
#pragma unroll
  for (int i = 0; i < 8; ++i) v[i] = 0.f;
  return v;
}

__device__ __forceinline__ v8f wmma_bf16(const Frag16& a, const Frag16& b, v8f c) {
  // D = A(16x32 bf16) * B(32x16 bf16) + C(16x16 f32)
  return __builtin_amdgcn_wmma_f32_16x16x32_bf16(
      /*neg_a=*/false, a.v, /*neg_b=*/false, b.v,
      /*c_mod=*/(short)0, c, /*reuse_a=*/false, /*reuse_b=*/false);
}

// Low 32 bits of a shared-aperture flat address ARE the LDS byte address
// (aperture mapping truncates addr[31:0] -> LDS offset).
__device__ __forceinline__ unsigned int lds_off(const void* p) {
  return (unsigned int)(unsigned long long)p;
}

// Per-lane async gather into LDS (tracked by ASYNCcnt).
__device__ __forceinline__ void async_load_b128(unsigned int lds, const void* gp) {
  asm volatile("global_load_async_to_lds_b128 %0, %1, off"
               :: "v"(lds), "v"((unsigned long long)gp) : "memory");
}
__device__ __forceinline__ void wait_asynccnt0() {
  asm volatile("s_wait_asynccnt 0" ::: "memory");
}

// ---------------------------------------------------------------------------
// TDM: DMA a 2D bf16 tile (tile_w x tile_h, row stride row_stride elements)
// from global to LDS, inserting 16B of LDS padding after every 64B of data
// (-> 40-halfword padded rows for a 32-element tile row). TENSORcnt tracked.
// ---------------------------------------------------------------------------
__device__ __forceinline__ void tdm_load_2d(unsigned int lds, const void* gp,
                                            int tensor_w, int tensor_h,
                                            int tile_w, int tile_h,
                                            int row_stride) {
  unsigned long long ga = (unsigned long long)gp;
  u32x4 g0;
  g0[0] = 1u;                                            // count=1 (valid), user
  g0[1] = lds;                                           // lds_addr
  g0[2] = (unsigned int)ga;                              // global_addr[31:0]
  g0[3] = (unsigned int)((ga >> 32) & 0x01FFFFFFu)       // global_addr[56:32]
        | (2u << 30);                                    // type=2 ("image")
  i32x8 g1;
  g1[0] = (1 << 16)                                      // data_size = 2 bytes
        | (1 << 20)                                      // pad_enable
        | (3 << 22)                                      // pad_interval: 16 DW (64B)
        | (3 << 25);                                     // pad_amount: 4 DW (16B)
  g1[1] = (tensor_w & 0xFFFF) << 16;                     // tensor_dim0[15:0]
  g1[2] = ((tensor_w >> 16) & 0xFFFF)                    // tensor_dim0[31:16]
        | ((tensor_h & 0xFFFF) << 16);                   // tensor_dim1[15:0]
  g1[3] = ((tensor_h >> 16) & 0xFFFF)                    // tensor_dim1[31:16]
        | (tile_w << 16);                                // tile_dim0
  g1[4] = tile_h & 0xFFFF;                               // tile_dim1 (tile_dim2=0)
  g1[5] = row_stride;                                    // tensor_dim0_stride[31:0]
  g1[6] = 0;                                             // stride hi / dim1_stride lo
  g1[7] = 0;
  i32x4 z4 = {0, 0, 0, 0};
#if __clang_major__ >= 23
  i32x8 z8 = {0, 0, 0, 0, 0, 0, 0, 0};
  __builtin_amdgcn_tensor_load_to_lds(g0, g1, z4, z4, z8, 0);
#else
  __builtin_amdgcn_tensor_load_to_lds(g0, g1, z4, z4, 0);
#endif
}

// ---------------------------------------------------------------------------
// fp32 -> bf16 elementwise convert
// ---------------------------------------------------------------------------
__global__ void cvt_f32_bf16(const float* __restrict__ in, bf16* __restrict__ out, int n) {
  int i = blockIdx.x * blockDim.x + threadIdx.x;
  int stride = gridDim.x * blockDim.x;
  for (; i < n; i += stride) out[i] = (bf16)in[i];
}

// ---------------------------------------------------------------------------
// W[K][N] fp32 -> Wt[N][K] bf16 (32x32 LDS tiles)
// ---------------------------------------------------------------------------
__global__ void transpose_cvt(const float* __restrict__ W, bf16* __restrict__ Wt,
                              int K, int N) {
  __shared__ float t[32][33];
  const int tx = threadIdx.x, ty = threadIdx.y;
#pragma unroll
  for (int j = 0; j < 4; ++j) {
    int r = blockIdx.y * 32 + ty + j * 8;   // k
    int c = blockIdx.x * 32 + tx;           // n
    t[ty + j * 8][tx] = W[(size_t)r * N + c];
  }
  __syncthreads();
#pragma unroll
  for (int j = 0; j < 4; ++j) {
    int n = blockIdx.x * 32 + ty + j * 8;
    int k = blockIdx.y * 32 + tx;
    Wt[(size_t)n * K + k] = (bf16)t[tx][ty + j * 8];
  }
}

// ---------------------------------------------------------------------------
// C[M][N] = A[M][K] * Bt[N][K]^T + bias   (bf16 inputs, f32 accumulate)
// Block tile 128x128, K-step 32, 8 waves each computing 32x64 (2x4 WMMA tiles).
// A/B tiles staged by TDM, double buffered so DMA overlaps WMMA compute.
// ---------------------------------------------------------------------------
template <typename OutT>
__global__ __launch_bounds__(256) void gemm_bf16_wmma(
    const bf16* __restrict__ A, const bf16* __restrict__ Bt,
    const float* __restrict__ bias, OutT* __restrict__ C,
    int M, int N, int K) {
  __shared__ bf16 As[2][128][40];  // 32 data + 8 pad halfwords per row
  __shared__ bf16 Bs[2][128][40];

  const int tid = threadIdx.x;
  const int lane = tid & 31, wid = tid >> 5;
  const int l16 = lane & 15, hi = lane >> 4;
  const int wm = wid & 3, wn = wid >> 2;   // 4x2 wave grid
  const int mbase = blockIdx.y * 128, nbase = blockIdx.x * 128;

  v8f acc[2][4];
#pragma unroll
  for (int i = 0; i < 2; ++i)
#pragma unroll
    for (int n = 0; n < 4; ++n) acc[i][n] = zero8();

  // Prologue: DMA tile 0.
  if (wid == 0) {
    tdm_load_2d(lds_off(&As[0][0][0]), &A[(size_t)mbase * K], K, M, 32, 128, K);
    tdm_load_2d(lds_off(&Bs[0][0][0]), &Bt[(size_t)nbase * K], K, N, 32, 128, K);
    __builtin_amdgcn_s_wait_tensorcnt(0);
  }
  __syncthreads();

  for (int kb = 0; kb < K; kb += 32) {
    const int buf = (kb >> 5) & 1;
    // Prefetch next tile into the other buffer while we compute this one.
    if (kb + 32 < K && wid == 0) {
      tdm_load_2d(lds_off(&As[buf ^ 1][0][0]), &A[(size_t)mbase * K + kb + 32],
                  K, M, 32, 128, K);
      tdm_load_2d(lds_off(&Bs[buf ^ 1][0][0]), &Bt[(size_t)nbase * K + kb + 32],
                  K, N, 32, 128, K);
    }

    Frag16 af[2];
#pragma unroll
    for (int i = 0; i < 2; ++i) {
      int r = wm * 32 + i * 16 + l16;
      int k0 = hi * 8;                 // A layout: lanes>=16 hold K+8
      af[i].h[0] = *(const v8bf*)&As[buf][r][k0];
      af[i].h[1] = *(const v8bf*)&As[buf][r][k0 + 16];
    }
#pragma unroll
    for (int n = 0; n < 4; ++n) {
      Frag16 bfr;
      int r = wn * 64 + n * 16 + l16;
      int k0 = hi * 8;                 // B layout: lanes>=16 hold K+8
      bfr.h[0] = *(const v8bf*)&Bs[buf][r][k0];
      bfr.h[1] = *(const v8bf*)&Bs[buf][r][k0 + 16];
#pragma unroll
      for (int i = 0; i < 2; ++i) acc[i][n] = wmma_bf16(af[i], bfr, acc[i][n]);
    }

    if (wid == 0) __builtin_amdgcn_s_wait_tensorcnt(0);
    __syncthreads();   // all reads of `buf` were consumed by WMMAs above
  }

#pragma unroll
  for (int i = 0; i < 2; ++i)
#pragma unroll
    for (int n = 0; n < 4; ++n) {
      int row = mbase + wm * 32 + i * 16 + hi * 8;  // C layout: VGPR r -> M=r(+8)
      int col = nbase + wn * 64 + n * 16 + l16;
      float bv = bias[col];
#pragma unroll
      for (int r = 0; r < 8; ++r) {
        float v = acc[i][n][r] + bv;
        C[(size_t)(row + r) * N + col] = (OutT)v;
      }
    }
}

// ---------------------------------------------------------------------------
// Fused dual-softmax flash attention.
//   Q,K,V: bf16 [B,S,D] (head h occupies columns h*64..h*64+63)
//   Fn,Fp: bf16 [B,S,D] feat_neg / feat_pos
// Block: 256 threads = 8 waves; each wave owns 16 query rows -> 128 q/block.
// K-tile staged via async-to-LDS; V transposed through registers into LDS.
// ---------------------------------------------------------------------------
__global__ __launch_bounds__(256) void attn_dual(
    const bf16* __restrict__ Q, const bf16* __restrict__ Kg,
    const bf16* __restrict__ Vg, bf16* __restrict__ Fn, bf16* __restrict__ Fp,
    int S, int D) {
  constexpr int HD = 64;
  __shared__ bf16 Ks[64][72];       // [key][d]
  __shared__ bf16 Vt[64][72];       // [d][key]
  __shared__ bf16 Pb[8][16][72];    // per-wave P staging (C-layout -> A-layout)

  const int tid = threadIdx.x;
  const int lane = tid & 31, wid = tid >> 5;
  const int l16 = lane & 15, hi = lane >> 4;
  const int b = blockIdx.z, h = blockIdx.y;
  const int q0 = blockIdx.x * 128 + wid * 16;

  // Q fragments held in registers for the whole kernel (2 x 16x32 over d=64).
  Frag16 qf[2];
  {
    const bf16* qp = Q + ((size_t)b * S + q0 + l16) * D + h * HD;
#pragma unroll
    for (int f = 0; f < 2; ++f) {
      int k0 = f * 32 + hi * 8;
      qf[f].h[0] = *(const v8bf*)(qp + k0);
      qf[f].h[1] = *(const v8bf*)(qp + k0 + 16);
    }
  }

  float mpos[8], mneg[8], lpos[8], lneg[8];
  v8f Op[4], On[4];
#pragma unroll
  for (int i = 0; i < 8; ++i) { mpos[i] = -3.0e38f; mneg[i] = -3.0e38f; lpos[i] = 0.f; lneg[i] = 0.f; }
#pragma unroll
  for (int t = 0; t < 4; ++t) { Op[t] = zero8(); On[t] = zero8(); }

  for (int kt = 0; kt < S / 64; ++kt) {
    __syncthreads();   // previous tile's LDS reads all consumed by WMMAs
    // Stage K tile (async DMA into LDS) and V tile (transposed via regs).
#pragma unroll
    for (int i = 0; i < 2; ++i) {
      int c = tid + i * 256;                // 512 chunks of 8 over 64x64
      int row = c >> 3, col = (c & 7) * 8;  // row=key, col=d
      const size_t g = ((size_t)b * S + kt * 64 + row) * D + h * HD + col;
      async_load_b128(lds_off(&Ks[row][col]), &Kg[g]);
      v8bf vv = *(const v8bf*)&Vg[g];
#pragma unroll
      for (int e = 0; e < 8; ++e) Vt[col + e][row] = vv[e];
    }
    wait_asynccnt0();   // K tile landed in LDS (Vt stores fenced by barrier)
    __syncthreads();

    // S-tile = Q * K^T (16 q-rows x 64 keys), scaled by 1/sqrt(64).
    v8f sa[4];
#pragma unroll
    for (int n = 0; n < 4; ++n) {
      sa[n] = zero8();
      int kr = n * 16 + l16;
      int d0 = hi * 8;
      Frag16 b0, b1;
      b0.h[0] = *(const v8bf*)&Ks[kr][d0];      b0.h[1] = *(const v8bf*)&Ks[kr][d0 + 16];
      b1.h[0] = *(const v8bf*)&Ks[kr][32 + d0]; b1.h[1] = *(const v8bf*)&Ks[kr][32 + d0 + 16];
      sa[n] = wmma_bf16(qf[0], b0, sa[n]);
      sa[n] = wmma_bf16(qf[1], b1, sa[n]);
#pragma unroll
      for (int i = 0; i < 8; ++i) sa[n][i] *= 0.125f;
    }

    // Dual online-softmax bookkeeping (per-row over 16-lane halves).
    float ap[8], an[8];
#pragma unroll
    for (int i = 0; i < 8; ++i) {
      float vmax = -3.4e38f, vmin = 3.4e38f;
#pragma unroll
      for (int n = 0; n < 4; ++n) { vmax = fmaxf(vmax, sa[n][i]); vmin = fminf(vmin, sa[n][i]); }
      for (int m = 1; m < 16; m <<= 1) {
        vmax = fmaxf(vmax, __shfl_xor(vmax, m, 16));
        vmin = fminf(vmin, __shfl_xor(vmin, m, 16));
      }
      float mp = fmaxf(mpos[i], vmax);
      float mn = fmaxf(mneg[i], -vmin);
      ap[i] = __expf(mpos[i] - mp);
      an[i] = __expf(mneg[i] - mn);
      mpos[i] = mp; mneg[i] = mn;
      lpos[i] *= ap[i]; lneg[i] *= an[i];
    }
#pragma unroll
    for (int t = 0; t < 4; ++t)
#pragma unroll
      for (int i = 0; i < 8; ++i) { Op[t][i] *= ap[i]; On[t][i] *= an[i]; }

#define ATTN_BRANCH(SGN, MARR, LARR, OARR)                                        \
    {                                                                             \
      float ps[8];                                                                \
      _Pragma("unroll") for (int i = 0; i < 8; ++i) ps[i] = 0.f;                  \
      _Pragma("unroll") for (int n = 0; n < 4; ++n)                               \
      _Pragma("unroll") for (int i = 0; i < 8; ++i) {                             \
        float p = __expf(SGN sa[n][i] - MARR[i]);                                 \
        ps[i] += p;                                                               \
        Pb[wid][i + hi * 8][n * 16 + l16] = (bf16)p;                              \
      }                                                                           \
      _Pragma("unroll") for (int i = 0; i < 8; ++i) {                             \
        for (int m = 1; m < 16; m <<= 1) ps[i] += __shfl_xor(ps[i], m, 16);       \
        LARR[i] += ps[i];                                                         \
      }                                                                           \
      asm volatile("s_wait_dscnt 0" ::: "memory");                                \
      _Pragma("unroll") for (int c = 0; c < 2; ++c) {                             \
        Frag16 pf;                                                                \
        int k0 = c * 32 + hi * 8;                                                 \
        pf.h[0] = *(const v8bf*)&Pb[wid][l16][k0];                                \
        pf.h[1] = *(const v8bf*)&Pb[wid][l16][k0 + 16];                           \
        _Pragma("unroll") for (int t = 0; t < 4; ++t) {                           \
          Frag16 vf;                                                              \
          int dr = t * 16 + l16;                                                  \
          int kk = c * 32 + hi * 8;                                               \
          vf.h[0] = *(const v8bf*)&Vt[dr][kk];                                    \
          vf.h[1] = *(const v8bf*)&Vt[dr][kk + 16];                               \
          OARR[t] = wmma_bf16(pf, vf, OARR[t]);                                   \
        }                                                                         \
      }                                                                           \
    }

    ATTN_BRANCH(+, mpos, lpos, Op)   // softmax(+logits)
    ATTN_BRANCH(-, mneg, lneg, On)   // softmax(-logits)
#undef ATTN_BRANCH
  }

  // Normalize and write feat (bf16, [B,S,D] layout for the output projection).
#pragma unroll
  for (int t = 0; t < 4; ++t)
#pragma unroll
    for (int i = 0; i < 8; ++i) {
      int row = q0 + i + hi * 8;
      int col = h * HD + t * 16 + l16;
      size_t o = ((size_t)b * S + row) * D + col;
      Fp[o] = (bf16)(Op[t][i] / lpos[i]);
      Fn[o] = (bf16)(On[t][i] / lneg[i]);
    }
}

// ---------------------------------------------------------------------------
// Host launcher
// ---------------------------------------------------------------------------
extern "C" void kernel_launch(void* const* d_in, const int* in_sizes, int n_in,
                              void* d_out, int out_size, void* d_ws, size_t ws_size,
                              hipStream_t stream) {
  (void)in_sizes; (void)n_in; (void)out_size; (void)ws_size;
  constexpr int B = 2, S = 2048, D = 1024, H = 16;
  constexpr size_t BSD = (size_t)B * S * D;   // 4M elements
  constexpr size_t DD = (size_t)D * D;        // 1M elements

  const float* tokens = (const float*)d_in[0];
  const float* Wq = (const float*)d_in[1];
  const float* bq = (const float*)d_in[2];
  const float* Wk = (const float*)d_in[3];
  const float* bk = (const float*)d_in[4];
  const float* Wv = (const float*)d_in[5];
  const float* bv = (const float*)d_in[6];
  const float* Wo = (const float*)d_in[7];
  const float* bo = (const float*)d_in[8];

  char* p = (char*)d_ws;                     // ~56 MB total
  bf16* tok = (bf16*)p;  p += BSD * 2;
  bf16* WqT = (bf16*)p;  p += DD * 2;
  bf16* WkT = (bf16*)p;  p += DD * 2;
  bf16* WvT = (bf16*)p;  p += DD * 2;
  bf16* WoT = (bf16*)p;  p += DD * 2;
  bf16* Qb  = (bf16*)p;  p += BSD * 2;
  bf16* Kb  = (bf16*)p;  p += BSD * 2;
  bf16* Vb  = (bf16*)p;  p += BSD * 2;
  bf16* Fn  = (bf16*)p;  p += BSD * 2;
  bf16* Fp  = (bf16*)p;  p += BSD * 2;

  float* outn = (float*)d_out;
  float* outp = outn + BSD;

  // 1) Precision conversion + weight transposes.
  cvt_f32_bf16<<<2048, 256, 0, stream>>>(tokens, tok, (int)BSD);
  dim3 tb(32, 8), tg(D / 32, D / 32);
  transpose_cvt<<<tg, tb, 0, stream>>>(Wq, WqT, D, D);
  transpose_cvt<<<tg, tb, 0, stream>>>(Wk, WkT, D, D);
  transpose_cvt<<<tg, tb, 0, stream>>>(Wv, WvT, D, D);
  transpose_cvt<<<tg, tb, 0, stream>>>(Wo, WoT, D, D);

  // 2) QKV projections (bf16 out).
  dim3 gg(D / 128, (B * S) / 128);
  gemm_bf16_wmma<bf16><<<gg, 256, 0, stream>>>(tok, WqT, bq, Qb, B * S, D, D);
  gemm_bf16_wmma<bf16><<<gg, 256, 0, stream>>>(tok, WkT, bk, Kb, B * S, D, D);
  gemm_bf16_wmma<bf16><<<gg, 256, 0, stream>>>(tok, WvT, bv, Vb, B * S, D, D);

  // 3) Fused dual-softmax attention.
  attn_dual<<<dim3(S / 128, H, B), 256, 0, stream>>>(Qb, Kb, Vb, Fn, Fp, S, D);

  // 4) Output projections (fp32 to d_out, neg then pos per reference order).
  gemm_bf16_wmma<float><<<gg, 256, 0, stream>>>(Fn, WoT, bo, outn, B * S, D, D);
  gemm_bf16_wmma<float><<<gg, 256, 0, stream>>>(Fp, WoT, bo, outp, B * S, D, D);
}
